// SDConv_direct_35665408426626
// MI455X (gfx1250) — matrix operand order repaired and checked by reading, not compiled, not run on hardware
//
#include <hip/hip_runtime.h>
#include <hip/hip_bf16.h>
#include <stdint.h>

typedef __attribute__((ext_vector_type(16))) _Float16 v16h;
typedef __attribute__((ext_vector_type(8)))  float    v8f;

#define BATCH 8
#define CIN   16
#define COUT  16
#define HDIM  224
#define WDIM  224
#define HW    (HDIM * WDIM)
#define WT    (WDIM / 16)              // 14 tiles along W
#define NTILES (BATCH * HDIM * WT)     // 25088 tiles of 16 pixels

// LDS weight layout for WMMA-B construction:
// wB[tap][v][n] = packed f16 pair ( W[n][2v][tap], W[n][2v+1][tap] ), tap 0..9 (tap 9 = zero pad)
#define WB_ENTRIES (10 * 8 * 16)

__global__ __launch_bounds__(256)
void sdconv_wmma_kernel(const float* __restrict__ U,
                        const float* __restrict__ V,
                        const float* __restrict__ Wt,
                        float* __restrict__ out)
{
    __shared__ uint32_t wB[WB_ENTRIES];          // 5 KB: weights in WMMA-B pair layout
    __shared__ float    xpose[8][16][17];        // per-wave 16x16 f32 transpose buffer (pad 17)

    const int tid = threadIdx.x;

    // ---- Stage weights into LDS in B-matrix pair-packed layout (once per block) ----
    for (int e = tid; e < WB_ENTRIES; e += 256) {
        const int tap = e >> 7;           // 0..9
        const int v   = (e >> 4) & 7;     // channel pair index
        const int n   = e & 15;           // output channel
        uint32_t pk = 0u;
        if (tap < 9) {
            const float w0 = Wt[n * 144 + (2 * v    ) * 9 + tap];
            const float w1 = Wt[n * 144 + (2 * v + 1) * 9 + tap];
            union { _Float16 h[2]; uint32_t u; } cv;
            cv.h[0] = (_Float16)w0;
            cv.h[1] = (_Float16)w1;
            pk = cv.u;
        }
        wB[e] = pk;
    }
    __syncthreads();

    const int wave = tid >> 5;   // 8 waves of 32 (wave32)
    const int lane = tid & 31;
    const int hh   = lane >> 4;  // lane half: selects K sub-runs & channel group
    const int ln   = lane & 15;

    // ---- Build B registers (held live across all tiles): 5 chunks of K=32 ----
    // B layout (16-bit, 32x16): lane half hh holds K=32q+8hh+[0..7] in elements 0-7, K=32q+16+8hh+[0..7] in 8-15;
    // element s: tap = 2q + (s>>3), c = 8hh + (s&7), with k = tap*16 + c (same order as A).
    v16h Breg[5];
    #pragma unroll
    for (int q = 0; q < 5; ++q) {
        const int tap = 2 * q + hh;
        #pragma unroll
        for (int v = 0; v < 8; ++v) {
            union { uint32_t u; _Float16 h[2]; } cv;
            cv.u = wB[((2 * q + (v >> 2)) * 8 + 4 * hh + (v & 3)) * 16 + ln];
            Breg[q][2 * v]     = cv.h[0];
            Breg[q][2 * v + 1] = cv.h[1];
        }
    }

    const int nWaves = gridDim.x * 8;
    const int gw     = blockIdx.x * 8 + wave;

    for (int tile = gw; tile < NTILES; tile += nWaves) {
        const int b   = tile / (HDIM * WT);
        const int rem = tile - b * (HDIM * WT);
        const int h   = rem / WT;
        const int w0  = (rem - h * WT) * 16;
        const int w   = w0 + ln;                 // this lane's pixel column

        const float* __restrict__ Ub = U + (size_t)b * CIN * HW;
        const float* __restrict__ Vb = V + (size_t)b * CIN * HW;
        const int cbase = hh * 8;                // this lane half owns channels cbase..cbase+7
        const int ctr   = h * WDIM + w;

        // Center values: one load per channel, reused by all 10 tap-runs of this tile.
        float uc[8], vc[8];
        #pragma unroll
        for (int j = 0; j < 8; ++j) {
            const int idx = (cbase + j) * HW + ctr;
            uc[j] = Ub[idx];
            vc[j] = Vb[idx];
        }

        v8f acc = {};
        #pragma unroll
        for (int q = 0; q < 5; ++q) {
            v16h A = {};                          // zero-init covers K=144..159 padding
            #pragma unroll
            for (int run = 0; run < 2; ++run) {
                const int tap = 2 * q + run;      // uniform across the whole wave
                if (tap < 9) {
                    const int di = tap / 3 - 1;
                    const int dj = tap % 3 - 1;
                    const int hn = h + di;
                    const int wn = w + dj;
                    const bool inb = (hn >= 0) & (hn < HDIM) & (wn >= 0) & (wn < WDIM);
                    const int nidx = hn * WDIM + wn;
                    #pragma unroll
                    for (int j = 0; j < 8; ++j) {
                        const int c = cbase + j;
                        const float un = inb ? Ub[c * HW + nidx] : 0.0f;
                        const float vn = inb ? Vb[c * HW + nidx] : 0.0f;
                        const float ud = un - uc[j];
                        const float vd = vn - vc[j];
                        const float s  = __builtin_amdgcn_rsqf(__builtin_fmaf(vd, vd, 1.0f));
                        // A layout: element e<8 -> K=32q+8hh+e (tap 2q, c=cbase+e);
                        //           e>=8     -> K=32q+16+8hh+(e-8) (tap 2q+1)
                        A[run * 8 + j] = (_Float16)(s * ud);
                    }
                }
            }
            // D = A(16x32 f16) x B(32x16 f16) + C(f32)
            acc = __builtin_amdgcn_wmma_f32_16x16x32_f16(
                false, A, false, Breg[q], (short)0, acc, false, false);
        }

        // ---- Transpose C through LDS for coalesced b128 stores ----
        // C layout: lane (hh,ln): VGPR r holds pixel M = hh*8 + r, channel N = ln.
        float* xp = &xpose[wave][0][0];
        #pragma unroll
        for (int r = 0; r < 8; ++r)
            xp[(hh * 8 + r) * 17 + ln] = acc[r];
        __builtin_amdgcn_wave_barrier();          // same-wave DS ordering (dscnt waits inserted)

        const int n2   = lane >> 1;               // output channel this lane stores
        const int part = lane & 1;                // which 8-pixel half
        float r0 = xp[(part * 8 + 0) * 17 + n2];
        float r1 = xp[(part * 8 + 1) * 17 + n2];
        float r2 = xp[(part * 8 + 2) * 17 + n2];
        float r3 = xp[(part * 8 + 3) * 17 + n2];
        float r4 = xp[(part * 8 + 4) * 17 + n2];
        float r5 = xp[(part * 8 + 5) * 17 + n2];
        float r6 = xp[(part * 8 + 6) * 17 + n2];
        float r7 = xp[(part * 8 + 7) * 17 + n2];
        __builtin_amdgcn_wave_barrier();

        float* op = out + ((size_t)(b * COUT + n2) * HDIM + h) * WDIM + w0 + part * 8;
        float4 lo = make_float4(r0, r1, r2, r3);
        float4 hi = make_float4(r4, r5, r6, r7);
        *(float4*)(op + 0) = lo;                  // 16B-aligned: w0 multiple of 16
        *(float4*)(op + 4) = hi;
    }
}

extern "C" void kernel_launch(void* const* d_in, const int* in_sizes, int n_in,
                              void* d_out, int out_size, void* d_ws, size_t ws_size,
                              hipStream_t stream) {
    const float* U  = (const float*)d_in[0];
    const float* V  = (const float*)d_in[1];
    const float* Wt = (const float*)d_in[2];
    float* out = (float*)d_out;

    // 25088 tiles; 784 blocks x 8 waves = 6272 waves -> 4 tiles per wave,
    // amortizing the 40-VGPR weight-matrix setup.
    sdconv_wmma_kernel<<<784, 256, 0, stream>>>(U, V, Wt, out);
}
